// Decoder_57913339019701
// MI455X (gfx1250) — matrix operand order repaired and checked by reading, not compiled, not run on hardware
//
#include <hip/hip_runtime.h>

typedef _Float16 half_t;
typedef __attribute__((ext_vector_type(8)))  _Float16 v8h;
typedef __attribute__((ext_vector_type(16))) _Float16 v16h;
typedef __attribute__((ext_vector_type(8)))  float    v8f;

// ---------------- problem constants ----------------
constexpr int NB = 16, NT = 256, NL = 400, NID = 512, NOD = 80, NDU = 1024, NPRE = 256;
constexpr int NATT = 128, NACH = 32, NAK = 31;
constexpr float ZPV = 0.1f, SCALINGV = 2.0f, EPSV = 1e-5f;
constexpr int KX  = NID + NPRE;   // 768  (x = [att_c | prenet])
constexpr int K0T = KX + NDU;     // 1792 (LSTM0 fused K)
constexpr int K1T = 2 * NDU;      // 2048 (LSTM1 fused K)
constexpr int NG  = 4 * NDU;      // 4096 gate columns
constexpr int NWG = 32;           // persistent grid
constexpr int TPAD = NL + 4;      // 404 (conv halo, pad=2 each side)
constexpr int CIP1 = 96;          // layer-1 Cin padded 80 -> 96

// ---------------- workspace layout (bytes) ----------------
constexpr size_t OFF_BAR  = 0;
constexpr size_t OFF_W0H  = 256;
constexpr size_t OFF_W1H  = OFF_W0H + (size_t)NG * K0T * 2;
constexpr size_t OFF_BS0  = OFF_W1H + (size_t)NG * K1T * 2;
constexpr size_t OFF_BS1  = OFF_BS0 + (size_t)NG * 4;
constexpr size_t OFF_ENCP = OFF_BS1 + (size_t)NG * 4;
constexpr size_t OFF_PR1  = OFF_ENCP + (size_t)NB * NT * NATT * 4;
constexpr size_t OFF_PRH  = OFF_PR1 + (size_t)NL * NB * NPRE * 4;
constexpr size_t OFF_A0   = OFF_PRH + (size_t)NL * NB * NPRE * 2;
constexpr size_t OFF_A1   = OFF_A0 + (size_t)NB * K0T * 2;
constexpr size_t OFF_Z0   = OFF_A1 + (size_t)NB * K1T * 2;
constexpr size_t OFF_C0   = OFF_Z0 + (size_t)NB * NDU * 4;
constexpr size_t OFF_Z1   = OFF_C0 + (size_t)NB * NDU * 4;
constexpr size_t OFF_C1   = OFF_Z1 + (size_t)NB * NDU * 4;
constexpr size_t OFF_G0   = OFF_C1 + (size_t)NB * NDU * 4;
constexpr size_t OFF_G1   = OFF_G0 + (size_t)NB * NG * 4;
constexpr size_t OFF_CUM  = OFF_G1 + (size_t)NB * NG * 4;
constexpr size_t OFF_DECP = OFF_CUM + (size_t)NB * NT * 4;
constexpr size_t OFF_ATTC = OFF_DECP + (size_t)NB * NATT * 4;
constexpr size_t OFF_XT0  = OFF_ATTC + (size_t)2 * NB * NID * 4;
constexpr size_t OFF_XTA  = OFF_XT0 + (size_t)NB * TPAD * CIP1 * 2;
constexpr size_t OFF_XTB  = OFF_XTA + (size_t)NB * TPAD * 512 * 2;
constexpr size_t OFF_PW1  = OFF_XTB + (size_t)NB * TPAD * 512 * 2;
constexpr size_t OFF_PW2  = OFF_PW1 + (size_t)512 * 5 * CIP1 * 2;
constexpr size_t OFF_PW3  = OFF_PW2 + (size_t)512 * 5 * 512 * 2;
constexpr size_t OFF_PW4  = OFF_PW3 + (size_t)512 * 5 * 512 * 2;
constexpr size_t OFF_PW5  = OFF_PW4 + (size_t)512 * 5 * 512 * 2;
constexpr size_t OFF_SC   = OFF_PW5 + (size_t)80 * 5 * 512 * 2;
constexpr size_t OFF_SH   = OFF_SC + (size_t)5 * 512 * 4;
constexpr size_t WS_TOTAL = OFF_SH + (size_t)5 * 512 * 4;

// d_out layout (floats): after | before | logits | att_ws
constexpr size_t DO_BEFORE = (size_t)NB * NL * NOD;
constexpr size_t DO_LOGITS = DO_BEFORE * 2;
constexpr size_t DO_ATTWS  = DO_LOGITS + (size_t)NB * NL;

// ---------------- helpers ----------------
__device__ __forceinline__ float sigf(float x) { return 1.0f / (1.0f + expf(-x)); }

// A fragment: 16xK tile, row-major source with row stride lda (halves).
// lane l<16 : row l, K {0..15}; lane l>=16 : row l-16, K {16..31}
__device__ __forceinline__ v16h load_afrag(const half_t* A, int lda, int k0) {
    int lane = threadIdx.x & 31;
    const half_t* p = A + (size_t)(lane & 15) * lda + k0 + ((lane >> 4) << 4);
    v8h lo = *(const v8h*)(p);
    v8h hi = *(const v8h*)(p + 8);
    return __builtin_shufflevector(lo, hi, 0,1,2,3,4,5,6,7,8,9,10,11,12,13,14,15);
}

__device__ __forceinline__ float block_max256(float v, float* s_red) {
    for (int off = 16; off; off >>= 1) v = fmaxf(v, __shfl_xor(v, off, 32));
    __syncthreads();
    if ((threadIdx.x & 31) == 0) s_red[threadIdx.x >> 5] = v;
    __syncthreads();
    float r = s_red[0];
#pragma unroll
    for (int i = 1; i < 8; ++i) r = fmaxf(r, s_red[i]);
    return r;
}
__device__ __forceinline__ float block_sum256(float v, float* s_red) {
    for (int off = 16; off; off >>= 1) v += __shfl_xor(v, off, 32);
    __syncthreads();
    if ((threadIdx.x & 31) == 0) s_red[threadIdx.x >> 5] = v;
    __syncthreads();
    float r = s_red[0];
#pragma unroll
    for (int i = 1; i < 8; ++i) r += s_red[i];
    return r;
}

// device-scope sense-reversing barrier over NWG resident workgroups
__device__ __forceinline__ void grid_barrier(int* bar) {
    __syncthreads();
    if (threadIdx.x == 0) {
        __threadfence();
        int gen = __hip_atomic_load(bar + 1, __ATOMIC_RELAXED, __HIP_MEMORY_SCOPE_AGENT);
        int arrived = __hip_atomic_fetch_add(bar, 1, __ATOMIC_ACQ_REL, __HIP_MEMORY_SCOPE_AGENT);
        if (arrived == NWG - 1) {
            __hip_atomic_store(bar, 0, __ATOMIC_RELAXED, __HIP_MEMORY_SCOPE_AGENT);
            __hip_atomic_store(bar + 1, gen + 1, __ATOMIC_RELEASE, __HIP_MEMORY_SCOPE_AGENT);
        } else {
            while (__hip_atomic_load(bar + 1, __ATOMIC_ACQUIRE, __HIP_MEMORY_SCOPE_AGENT) == gen) {
                __builtin_amdgcn_s_sleep(2);
            }
        }
    }
    __syncthreads();
}

// 16xNG gate GEMM: G[16][4096] (=A[16][K] * W[4096][K]^T), one 16-col tile per wave
__device__ __forceinline__ void wmma_gates(const half_t* A, const half_t* Wm, int K,
                                           float* G, int wave) {
    int lane = threadIdx.x & 31;
    int n0 = wave << 4;
    const half_t* arow = A + (size_t)(lane & 15) * K + ((lane >> 4) << 4);
    const half_t* wrow = Wm + (size_t)(n0 + (lane & 15)) * K + ((lane >> 4) << 4);
    v8f acc = {};
    for (int k0 = 0; k0 < K; k0 += 32) {
        v8h alo = *(const v8h*)(arow + k0);
        v8h ahi = *(const v8h*)(arow + k0 + 8);
        v16h af = __builtin_shufflevector(alo, ahi, 0,1,2,3,4,5,6,7,8,9,10,11,12,13,14,15);
        v16h bf = *(const v16h*)(wrow + k0);
        // WGP-scope prefetch of the upcoming weight chunk (locality 3 -> nearest cache;
        // SYS/DEV-scope prefetch would stop at GL2, where weights already live).
        __builtin_prefetch(wrow + k0 + 256, 0, 3);
        acc = __builtin_amdgcn_wmma_f32_16x16x32_f16(false, af, false, bf, (short)0, acc,
                                                     false, false);
    }
    int n = n0 + (lane & 15);
    int mb = (lane >> 4) << 3;
#pragma unroll
    for (int r = 0; r < 8; ++r) G[(size_t)(mb + r) * NG + n] = acc[r];
}

// ---------------- prep kernels ----------------
__global__ void prep_lstm(const float* wih0, const float* whh0, const float* bih0,
                          const float* bhh0, const float* wih1, const float* whh1,
                          const float* bih1, const float* bhh1,
                          half_t* W0H, half_t* W1H, float* BS0, float* BS1) {
    size_t stride = (size_t)gridDim.x * blockDim.x;
    size_t t0 = (size_t)blockIdx.x * blockDim.x + threadIdx.x;
    for (size_t i = t0; i < (size_t)NG * K0T; i += stride) {
        int n = (int)(i / K0T), k = (int)(i % K0T);
        float v = (k < KX) ? wih0[(size_t)n * KX + k] : whh0[(size_t)n * NDU + (k - KX)];
        W0H[i] = (half_t)v;
    }
    for (size_t i = t0; i < (size_t)NG * K1T; i += stride) {
        int n = (int)(i / K1T), k = (int)(i % K1T);
        float v = (k < NDU) ? wih1[(size_t)n * NDU + k] : whh1[(size_t)n * NDU + (k - NDU)];
        W1H[i] = (half_t)v;
    }
    for (size_t i = t0; i < (size_t)NG; i += stride) {
        BS0[i] = bih0[i] + bhh0[i];
        BS1[i] = bih1[i] + bhh1[i];
    }
}

__global__ void prep_encproj(const float* hs, const float* W_enc, const float* b_enc,
                             float* ENCP) {
    size_t i = (size_t)blockIdx.x * blockDim.x + threadIdx.x;  // over NB*NT*NATT
    if (i >= (size_t)NB * NT * NATT) return;
    int a = (int)(i % NATT);
    size_t bt = i / NATT;
    const float* hrow = hs + bt * NID;
    float acc = b_enc[a];
    for (int k = 0; k < NID; ++k) acc += hrow[k] * W_enc[(size_t)k * NATT + a];
    ENCP[i] = acc;
}

__global__ void prep_prenet1(const float* ys, const float* p_w1, const float* p_b1,
                             float* pr1) {
    size_t i = (size_t)blockIdx.x * blockDim.x + threadIdx.x;  // [l][b][j]
    if (i >= (size_t)NL * NB * NPRE) return;
    int j = (int)(i % NPRE);
    size_t lb = i / NPRE;
    int b = (int)(lb % NB), l = (int)(lb / NB);
    float acc = p_b1[j];
    if (l > 0) {
        const float* y = ys + ((size_t)b * NL + (l - 1)) * NOD;
        const float* w = p_w1 + (size_t)j * NOD;
        for (int od = 0; od < NOD; ++od) acc += y[od] * w[od];
    }
    pr1[i] = fmaxf(acc, 0.0f);
}

__global__ void prep_prenet2(const float* pr1, const float* p_w2, const float* p_b2,
                             half_t* prh) {
    size_t i = (size_t)blockIdx.x * blockDim.x + threadIdx.x;
    if (i >= (size_t)NL * NB * NPRE) return;
    int j = (int)(i % NPRE);
    const float* x = pr1 + (i / NPRE) * NPRE;
    const float* w = p_w2 + (size_t)j * NPRE;
    float acc = p_b2[j];
    for (int k = 0; k < NPRE; ++k) acc += x[k] * w[k];
    prh[i] = (half_t)fmaxf(acc, 0.0f);
}

__global__ void prep_post(const float* w, int cout, int cin, int cip,
                          const float* gamma, const float* beta, const float* mean,
                          const float* var, half_t* wpad, float* scale, float* shift) {
    size_t i = (size_t)blockIdx.x * blockDim.x + threadIdx.x;
    size_t total = (size_t)cout * cip * 5;
    if (i < total) {
        int co = (int)(i / ((size_t)cip * 5));
        int rem = (int)(i % ((size_t)cip * 5));
        int kr = rem / cip, ci = rem % cip;
        float v = (ci < cin) ? w[((size_t)co * cin + ci) * 5 + kr] : 0.0f;
        wpad[(size_t)co * (5 * cip) + (size_t)kr * cip + ci] = (half_t)v;
    }
    if (i < (size_t)cout) {
        float sc = gamma[i] * rsqrtf(var[i] + EPSV);
        scale[i] = sc;
        shift[i] = beta[i] - mean[i] * sc;
    }
}

// ---------------- persistent decoder ----------------
__global__ void __launch_bounds__(256)
decoder_kernel(char* ws, const float* hs, const int* hlens, const float* W_dec,
               const float* loc_w, const float* W_att, const float* gvec_w,
               const float* gvec_b, const float* feat_w, const float* prob_w,
               const float* prob_b, float* dout) {
    int* bar = (int*)(ws + OFF_BAR);
    const half_t* W0H = (const half_t*)(ws + OFF_W0H);
    const half_t* W1H = (const half_t*)(ws + OFF_W1H);
    const float* BS0 = (const float*)(ws + OFF_BS0);
    const float* BS1 = (const float*)(ws + OFF_BS1);
    const float* ENCP = (const float*)(ws + OFF_ENCP);
    const half_t* PRH = (const half_t*)(ws + OFF_PRH);
    half_t* A0h = (half_t*)(ws + OFF_A0);
    half_t* A1h = (half_t*)(ws + OFF_A1);
    float* Z0 = (float*)(ws + OFF_Z0);
    float* C0 = (float*)(ws + OFF_C0);
    float* Z1 = (float*)(ws + OFF_Z1);
    float* C1 = (float*)(ws + OFF_C1);
    float* G0 = (float*)(ws + OFF_G0);
    float* G1 = (float*)(ws + OFF_G1);
    float* CUM = (float*)(ws + OFF_CUM);
    float* DECP = (float*)(ws + OFF_DECP);
    float* ATTC = (float*)(ws + OFF_ATTC);
    half_t* XT0h = (half_t*)(ws + OFF_XT0);
    float* out_before = dout + DO_BEFORE;
    float* out_logits = dout + DO_LOGITS;
    float* out_attws = dout + DO_ATTWS;

    const int wg = blockIdx.x;
    const int tid = threadIdx.x;
    const int gtid = wg * 256 + tid;
    const int myb = wg >> 1;
    const bool att_wg = (wg & 1) == 0;
    const int wave = (wg << 3) + (tid >> 5);
    const float gvb = gvec_b[0];

    __shared__ float s_lw[NACH * NAK];
    __shared__ float s_watt[NACH * NATT];
    __shared__ float s_gvec[NATT];
    __shared__ float s_decp[NATT];
    __shared__ float s_w[NT];
    __shared__ float s_red[8];

    for (int i = tid; i < NACH * NAK; i += 256) s_lw[i] = loc_w[i];
    for (int i = tid; i < NACH * NATT; i += 256) s_watt[i] = W_att[i];
    if (tid < NATT) s_gvec[tid] = gvec_w[tid];
    __syncthreads();

    for (int s = 0; s <= NL; ++s) {
        // ---- P_local: per-batch-row work, no cross-WG deps ----
        if (att_wg) {
            if (s < NL) {
                const int b = myb;
                const int hl = hlens[b];
                const int t = tid;  // 0..255
                if (tid < NATT) s_decp[tid] = DECP[b * NATT + tid];
                __syncthreads();
                const float* cw = CUM + b * NT;
                float pwv[NAK];
#pragma unroll
                for (int k = 0; k < NAK; ++k) {
                    int tt = t + k - (NAK - 1) / 2;
                    float v = 0.0f;
                    if (tt >= 0 && tt < NT)
                        v = (s == 0) ? ((tt < hl) ? 1.0f / (float)hl : 0.0f) : cw[tt];
                    pwv[k] = v;
                }
                float conv[NACH];
#pragma unroll
                for (int c = 0; c < NACH; ++c) {
                    float a = 0.0f;
#pragma unroll
                    for (int k = 0; k < NAK; ++k) a += s_lw[c * NAK + k] * pwv[k];
                    conv[c] = a;
                }
                const float* ep = ENCP + ((size_t)b * NT + t) * NATT;
                float ev = 0.0f;
                for (int a = 0; a < NATT; ++a) {
                    float acc = ep[a] + s_decp[a];
#pragma unroll
                    for (int c = 0; c < NACH; ++c) acc += conv[c] * s_watt[c * NATT + a];
                    ev += s_gvec[a] * tanhf(acc);
                }
                ev += gvb;
                float em = (t < hl) ? (SCALINGV * ev) : -1e30f;
                float m = block_max256(em, s_red);
                float ex = expf(em - m);
                float sum = block_sum256(ex, s_red);
                float w = ex / sum;
                s_w[t] = w;
                out_attws[((size_t)b * NL + s) * NT + t] = w;
                CUM[b * NT + t] = ((s == 0) ? 0.0f : cw[t]) + w;
                __syncthreads();
                // att_c = w @ hs[b]
                float* attc = ATTC + (size_t)(s & 1) * NB * NID + (size_t)b * NID;
                for (int d = tid; d < NID; d += 256) {
                    float acc = 0.0f;
                    const float* hb = hs + (size_t)b * NT * NID + d;
                    for (int tt = 0; tt < NT; ++tt) acc += s_w[tt] * hb[(size_t)tt * NID];
                    attc[d] = acc;
                    A0h[(size_t)b * K0T + d] = (half_t)acc;
                }
                A0h[(size_t)b * K0T + NID + tid] = PRH[((size_t)s * NB + b) * NPRE + tid];
            }
        } else {
            if (s >= 1) {
                const int b = myb;
                const int sp = s - 1;
                const float* z1b = Z1 + (size_t)b * NDU;
                const float* acb = ATTC + (size_t)(sp & 1) * NB * NID + (size_t)b * NID;
                if (tid < NOD) {
                    const float* fw = feat_w + (size_t)tid * (NDU + NID);
                    float acc = 0.0f;
                    for (int j = 0; j < NDU; ++j) acc += z1b[j] * fw[j];
                    for (int d = 0; d < NID; ++d) acc += acb[d] * fw[NDU + d];
                    out_before[((size_t)b * NL + sp) * NOD + tid] = acc;
                    XT0h[((size_t)b * TPAD + 2 + sp) * CIP1 + tid] = (half_t)acc;
                } else if (tid == NOD) {
                    float acc = prob_b[0];
                    for (int j = 0; j < NDU; ++j) acc += z1b[j] * prob_w[j];
                    for (int d = 0; d < NID; ++d) acc += acb[d] * prob_w[NDU + d];
                    out_logits[(size_t)b * NL + sp] = acc;
                }
            }
        }
        if (s == NL) break;

        grid_barrier(bar);  // A: x_h ready
        wmma_gates(A0h, W0H, K0T, G0, wave);
        grid_barrier(bar);  // B: gates0 ready

        // ---- EW0: LSTM0 cell + zoneout ----
        for (int e = gtid; e < NB * NDU; e += NWG * 256) {
            int b = e >> 10, j = e & (NDU - 1);
            const float* g = G0 + (size_t)b * NG;
            float gi = g[j] + BS0[j];
            float gf = g[NDU + j] + BS0[NDU + j];
            float gg = g[2 * NDU + j] + BS0[2 * NDU + j];
            float go = g[3 * NDU + j] + BS0[3 * NDU + j];
            float co = C0[e], ho = Z0[e];
            float cn = sigf(gf) * co + sigf(gi) * tanhf(gg);
            float hn = sigf(go) * tanhf(cn);
            float z = ZPV * ho + (1.0f - ZPV) * hn;
            float cc = ZPV * co + (1.0f - ZPV) * cn;
            Z0[e] = z;
            C0[e] = cc;
            half_t zh = (half_t)z;
            A1h[(size_t)b * K1T + j] = zh;          // LSTM1 x-part (this step)
            A0h[(size_t)b * K0T + KX + j] = zh;     // LSTM0 h-part (next step)
        }
        grid_barrier(bar);  // C: z0 ready
        wmma_gates(A1h, W1H, K1T, G1, wave);
        grid_barrier(bar);  // D: gates1 ready

        // ---- EW1: LSTM1 cell + zoneout, plus dec_proj for next step ----
        for (int e = gtid; e < NB * NDU; e += NWG * 256) {
            int b = e >> 10, j = e & (NDU - 1);
            const float* g = G1 + (size_t)b * NG;
            float gi = g[j] + BS1[j];
            float gf = g[NDU + j] + BS1[NDU + j];
            float gg = g[2 * NDU + j] + BS1[2 * NDU + j];
            float go = g[3 * NDU + j] + BS1[3 * NDU + j];
            float co = C1[e], ho = Z1[e];
            float cn = sigf(gf) * co + sigf(gi) * tanhf(gg);
            float hn = sigf(go) * tanhf(cn);
            float z = ZPV * ho + (1.0f - ZPV) * hn;
            float cc = ZPV * co + (1.0f - ZPV) * cn;
            Z1[e] = z;
            C1[e] = cc;
            A1h[(size_t)b * K1T + NDU + j] = (half_t)z;  // LSTM1 h-part (next step)
        }
        if (gtid < NB * NATT) {
            int b = gtid >> 7, a = gtid & (NATT - 1);
            const float* z0b = Z0 + (size_t)b * NDU;
            float acc = 0.0f;
            for (int j = 0; j < NDU; ++j) acc += z0b[j] * W_dec[(size_t)j * NATT + a];
            DECP[gtid] = acc;
        }
        grid_barrier(bar);  // E: z1/decp/A ready for next step
    }
}

// ---------------- postnet: conv1d(K=5)+BN(+tanh) as WMMA GEMM ----------------
__global__ void __launch_bounds__(256)
postnet_layer(const half_t* X, int cip, const half_t* W, const float* scale,
              const float* shift, half_t* Y, const float* before, float* after_out,
              int cout, int do_tanh, int is_final) {
    int wave = blockIdx.x * 8 + (threadIdx.x >> 5);
    int mtiles = cout >> 4;
    int total = mtiles * (NB * (NL / 16));
    if (wave >= total) return;
    int mt = wave % mtiles;
    int nt = wave / mtiles;
    int b = nt / (NL / 16);
    int t0 = (nt % (NL / 16)) << 4;
    int lane = threadIdx.x & 31;
    int lda = 5 * cip;
    const half_t* wbase = W + (size_t)(mt << 4) * lda;
    v8f acc = {};
    for (int kr = 0; kr < 5; ++kr) {
        const half_t* xb = X + ((size_t)b * TPAD + t0 + kr) * cip;  // pad=2 baked into layout
        const half_t* xlane = xb + (size_t)(lane & 15) * cip + ((lane >> 4) << 4);
        for (int kc = 0; kc < cip; kc += 32) {
            v16h af = load_afrag(wbase, lda, kr * cip + kc);
            v16h bf = *(const v16h*)(xlane + kc);
            __builtin_prefetch(xlane + kc + 128, 0, 3);  // WGP-scope prefetch
            acc = __builtin_amdgcn_wmma_f32_16x16x32_f16(false, af, false, bf, (short)0,
                                                         acc, false, false);
        }
    }
    int t = t0 + (lane & 15);
    int mb = (lane >> 4) << 3;
#pragma unroll
    for (int r = 0; r < 8; ++r) {
        int co = (mt << 4) + mb + r;
        float v = acc[r] * scale[co] + shift[co];
        if (do_tanh) v = tanhf(v);
        if (is_final) {
            size_t oi = ((size_t)b * NL + t) * NOD + co;
            after_out[oi] = v + before[oi];
        } else {
            Y[((size_t)b * TPAD + 2 + t) * 512 + co] = (half_t)v;
        }
    }
}

// ---------------- host launcher ----------------
extern "C" void kernel_launch(void* const* d_in, const int* in_sizes, int n_in,
                              void* d_out, int out_size, void* d_ws, size_t ws_size,
                              hipStream_t stream) {
    const float* hs = (const float*)d_in[0];
    const int* hlens = (const int*)d_in[1];
    const float* ys = (const float*)d_in[2];
    const float* W_enc = (const float*)d_in[3];
    const float* b_enc = (const float*)d_in[4];
    const float* W_dec = (const float*)d_in[5];
    const float* loc_w = (const float*)d_in[6];
    const float* W_att = (const float*)d_in[7];
    const float* gvec_w = (const float*)d_in[8];
    const float* gvec_b = (const float*)d_in[9];
    const float* p_w1 = (const float*)d_in[10];
    const float* p_b1 = (const float*)d_in[11];
    const float* p_w2 = (const float*)d_in[12];
    const float* p_b2 = (const float*)d_in[13];
    const float* wih0 = (const float*)d_in[14];
    const float* whh0 = (const float*)d_in[15];
    const float* bih0 = (const float*)d_in[16];
    const float* bhh0 = (const float*)d_in[17];
    const float* wih1 = (const float*)d_in[18];
    const float* whh1 = (const float*)d_in[19];
    const float* bih1 = (const float*)d_in[20];
    const float* bhh1 = (const float*)d_in[21];
    const float* feat_w = (const float*)d_in[22];
    const float* prob_w = (const float*)d_in[23];
    const float* prob_b = (const float*)d_in[24];

    char* ws = (char*)d_ws;
    float* dout = (float*)d_out;

    hipMemsetAsync(d_ws, 0, WS_TOTAL < ws_size ? WS_TOTAL : ws_size, stream);

    prep_lstm<<<2048, 256, 0, stream>>>(wih0, whh0, bih0, bhh0, wih1, whh1, bih1, bhh1,
                                        (half_t*)(ws + OFF_W0H), (half_t*)(ws + OFF_W1H),
                                        (float*)(ws + OFF_BS0), (float*)(ws + OFF_BS1));
    prep_encproj<<<2048, 256, 0, stream>>>(hs, W_enc, b_enc, (float*)(ws + OFF_ENCP));
    prep_prenet1<<<6400, 256, 0, stream>>>(ys, p_w1, p_b1, (float*)(ws + OFF_PR1));
    prep_prenet2<<<6400, 256, 0, stream>>>((const float*)(ws + OFF_PR1), p_w2, p_b2,
                                           (half_t*)(ws + OFF_PRH));

    const size_t pw_off[5] = {OFF_PW1, OFF_PW2, OFF_PW3, OFF_PW4, OFF_PW5};
    const int cins[5] = {NOD, 512, 512, 512, 512};
    const int couts[5] = {512, 512, 512, 512, NOD};
    const int cips[5] = {CIP1, 512, 512, 512, 512};
    for (int i = 0; i < 5; ++i) {
        prep_post<<<5120, 256, 0, stream>>>(
            (const float*)d_in[25 + i], couts[i], cins[i], cips[i],
            (const float*)d_in[30 + i], (const float*)d_in[35 + i],
            (const float*)d_in[40 + i], (const float*)d_in[45 + i],
            (half_t*)(ws + pw_off[i]), (float*)(ws + OFF_SC) + i * 512,
            (float*)(ws + OFF_SH) + i * 512);
    }

    decoder_kernel<<<NWG, 256, 0, stream>>>(ws, hs, hlens, W_dec, loc_w, W_att, gvec_w,
                                            gvec_b, feat_w, prob_w, prob_b, dout);

    const size_t xoff[6] = {OFF_XT0, OFF_XTA, OFF_XTB, OFF_XTA, OFF_XTB, 0};
    for (int i = 0; i < 5; ++i) {
        int blocks = ((couts[i] >> 4) * NB * (NL / 16) + 7) / 8;
        postnet_layer<<<blocks, 256, 0, stream>>>(
            (const half_t*)(ws + xoff[i]), cips[i], (const half_t*)(ws + pw_off[i]),
            (const float*)(ws + OFF_SC) + i * 512, (const float*)(ws + OFF_SH) + i * 512,
            (half_t*)(ws + xoff[i + 1]), dout + DO_BEFORE, dout, couts[i],
            (i < 4) ? 1 : 0, (i == 4) ? 1 : 0);
    }
}
